// TransformerBlock_20976620274098
// MI455X (gfx1250) — matrix operand order repaired and checked by reading, not compiled, not run on hardware
//
#include <hip/hip_runtime.h>
#include <hip/hip_bf16.h>
#include <cstdint>

// ---------------------------------------------------------------------------
// Problem constants
// ---------------------------------------------------------------------------
constexpr int Dm = 768;     // model dim
constexpr int Lm = 512;     // sequence length
constexpr int Hm = 12;      // heads
constexpr int DK = 64;      // head dim
constexpr int Bm = 8;       // batch
constexpr int Mtot = Bm * Lm;           // 4096 rows for all big GEMMs
constexpr float INV_TEMP = 1.0f / (8.0f + 1e-6f);  // 1/(sqrt(64)+eps)

typedef __attribute__((ext_vector_type(16))) __bf16 v16bf;
typedef __attribute__((ext_vector_type(8)))  __bf16 v8bf;
typedef __attribute__((ext_vector_type(4)))  __bf16 v4bf;
typedef __attribute__((ext_vector_type(8)))  float  v8f;
typedef __attribute__((ext_vector_type(4)))  int    v4i;

union FragBF { v16bf v; v8bf h[2]; };

// ---------------------------------------------------------------------------
// Async global->LDS staging (gfx1250 GLOBAL_LOAD_ASYNC_TO_LDS_B128, ASYNCcnt)
// with a guaranteed-compiling fallback through VGPRs.
// Probe result (round 2 diagnostic): builtin exists and takes v4i* params.
// ---------------------------------------------------------------------------
#if defined(__has_builtin)
#  if __has_builtin(__builtin_amdgcn_global_load_async_to_lds_b128)
#    define ASYNC_G2L 1
#  else
#    define ASYNC_G2L 0
#  endif
#else
#  define ASYNC_G2L 0
#endif

__device__ __forceinline__ void g2l_copy16(void* lds_dst, const void* gsrc) {
#if ASYNC_G2L
    __builtin_amdgcn_global_load_async_to_lds_b128(
        (v4i*)const_cast<void*>(gsrc), (v4i*)lds_dst, 0, 0);
#else
    *(v8bf*)lds_dst = *(const v8bf*)gsrc;
#endif
}

__device__ __forceinline__ void g2l_join() {
#if ASYNC_G2L
#  if __has_builtin(__builtin_amdgcn_s_wait_asynccnt)
    __builtin_amdgcn_s_wait_asynccnt(0);
#  else
    asm volatile("s_wait_asynccnt 0x0" ::: "memory");
#  endif
#endif
}

// ---------------------------------------------------------------------------
// f32 -> bf16 conversion (vectorized: float4 -> 4 x bf16); n must be mult of 4
// ---------------------------------------------------------------------------
__global__ void cvt_f32_bf16(const float* __restrict__ in, __bf16* __restrict__ out, int n4) {
    int i = blockIdx.x * blockDim.x + threadIdx.x;
    if (i < n4) {
        float4 v = ((const float4*)in)[i];
        v4bf r;
        r[0] = (__bf16)v.x; r[1] = (__bf16)v.y; r[2] = (__bf16)v.z; r[3] = (__bf16)v.w;
        ((v4bf*)out)[i] = r;
    }
}

// ---------------------------------------------------------------------------
// Tiled bf16 GEMM with WMMA (f32 accumulate).
// Block tile 128x64, BK=64 per stage, 256 threads = 8 waves in a 4x2 grid,
// each wave computes a 32x32 tile (4 accumulators) -> 8 WMMAs per stage with
// 8 fragment loads (1:1) and 12 double-barrier stages for K=768.
// MODE: 0 = store bf16
//       1 = relu(acc + bias) -> bf16
//       2 = acc + resid(f32) -> f32 out AND bf16 out
//       3 = acc + bias + resid(f32) -> f32 out
// ---------------------------------------------------------------------------
template<int MODE>
__global__ __launch_bounds__(256) void gemm_bf16_wmma(
    const __bf16* __restrict__ A, const __bf16* __restrict__ Bmat,
    const float*  __restrict__ bias, const float* __restrict__ resid,
    float* __restrict__ outF, __bf16* __restrict__ outB,
    int Md, int Nd, int Kd)
{
    __shared__ __bf16 As[128][72];   // row-major [m][k], stride 72
    __shared__ __bf16 Bs[64][72];    // B-column-major [n][k], stride 72

    const int t    = threadIdx.x;
    const int lane = t & 31;
    const int w    = t >> 5;
    const int gm0  = blockIdx.x * 128;
    const int gn0  = blockIdx.y * 64;
    const int wm   = (w & 3) * 32;
    const int wn   = (w >> 2) * 32;
    const int hf   = lane >> 4;
    const int l16  = lane & 15;

    v8f c00 = {}, c01 = {}, c10 = {}, c11 = {};

    for (int k0 = 0; k0 < Kd; k0 += 64) {
        // stage A tile 128x64: 1024 16B runs, 4 per thread (async DMA to LDS)
#pragma unroll
        for (int it = 0; it < 4; ++it) {
            int i = t + it * 256;
            int r = i >> 3, c8 = (i & 7) << 3;
            g2l_copy16(&As[r][c8], &A[(size_t)(gm0 + r) * Kd + k0 + c8]);
        }
        // stage B tile transposed: Bs[n][kk] = B[k0+kk][gn0+n] (VGPR scatter)
#pragma unroll
        for (int it = 0; it < 2; ++it) {
            int i = t + it * 256;
            int kk = i >> 3, n8 = (i & 7) << 3;
            v8bf vv = *(const v8bf*)&Bmat[(size_t)(k0 + kk) * Nd + gn0 + n8];
#pragma unroll
            for (int j = 0; j < 8; ++j) Bs[n8 + j][kk] = vv[j];
        }
        if (k0 + 64 < Kd)  // speculative prefetch of next B panel
            __builtin_prefetch(&Bmat[(size_t)(k0 + 64 + (t >> 3)) * Nd + gn0 + ((t & 7) << 3)], 0, 0);
        g2l_join();
        __syncthreads();

#pragma unroll
        for (int kc = 0; kc < 64; kc += 32) {
            FragBF a0, a1, b0, b1;
            int m = wm + l16;
            int n = wn + l16;
            a0.h[0] = *(const v8bf*)&As[m][kc + hf * 8];
            a0.h[1] = *(const v8bf*)&As[m][kc + 16 + hf * 8];
            a1.h[0] = *(const v8bf*)&As[m + 16][kc + hf * 8];
            a1.h[1] = *(const v8bf*)&As[m + 16][kc + 16 + hf * 8];
            b0.h[0] = *(const v8bf*)&Bs[n][kc + hf * 8];
            b0.h[1] = *(const v8bf*)&Bs[n][kc + 16 + hf * 8];
            b1.h[0] = *(const v8bf*)&Bs[n + 16][kc + hf * 8];
            b1.h[1] = *(const v8bf*)&Bs[n + 16][kc + 16 + hf * 8];

            c00 = __builtin_amdgcn_wmma_f32_16x16x32_bf16(false, a0.v, false, b0.v, (short)0, c00, false, false);
            c01 = __builtin_amdgcn_wmma_f32_16x16x32_bf16(false, a0.v, false, b1.v, (short)0, c01, false, false);
            c10 = __builtin_amdgcn_wmma_f32_16x16x32_bf16(false, a1.v, false, b0.v, (short)0, c10, false, false);
            c11 = __builtin_amdgcn_wmma_f32_16x16x32_bf16(false, a1.v, false, b1.v, (short)0, c11, false, false);
        }
        __syncthreads();
    }

    // Epilogue. C/D layout: row = vgpr + 8*(lane/16), col = lane%16
    auto emit = [&](const v8f& cc, int roff, int coff) {
#pragma unroll
        for (int vr = 0; vr < 8; ++vr) {
            int row = gm0 + wm + roff + vr + 8 * hf;
            int col = gn0 + wn + coff + l16;
            size_t i0 = (size_t)row * Nd + col;
            float x = cc[vr];
            if constexpr (MODE == 1) x = fmaxf(x + bias[col], 0.0f);
            if constexpr (MODE == 2) x += resid[i0];
            if constexpr (MODE == 3) x += bias[col] + resid[i0];
            if constexpr (MODE == 0 || MODE == 1) outB[i0] = (__bf16)x;
            if constexpr (MODE == 2) { outF[i0] = x; outB[i0] = (__bf16)x; }
            if constexpr (MODE == 3) outF[i0] = x;
        }
    };
    emit(c00, 0, 0); emit(c01, 0, 16); emit(c10, 16, 0); emit(c11, 16, 16);
}

// ---------------------------------------------------------------------------
// Fused dual-softmax attention (one block = (b, h, 32 query rows)).
// Full 32x512 score panel lives in LDS -> exact single-pass dual softmax,
// then P_pos/P_neg (bf16, LDS) x V via WMMA. 256 threads = 8 waves.
// ---------------------------------------------------------------------------
__global__ __launch_bounds__(256) void attn_dual_softmax(
    const __bf16* __restrict__ qproj, const __bf16* __restrict__ kproj,
    const __bf16* __restrict__ vproj,
    const float* __restrict__ Wpos, const float* __restrict__ Wneg,
    __bf16* __restrict__ Vp, __bf16* __restrict__ Vn)
{
    extern __shared__ char smem[];
    float*  Sf    = (float*)(smem);               // 32*512*4 = 65536
    __bf16* Pp    = (__bf16*)(smem + 65536);      // 32*512*2 = 32768
    __bf16* Pn    = (__bf16*)(smem + 98304);      // 32768
    __bf16* Qs    = (__bf16*)(smem + 131072);     // 32*72*2  = 4608
    __bf16* Ks    = (__bf16*)(smem + 135680);     // 64*72*2  = 9216
    __bf16* Vs    = (__bf16*)(smem + 144896);     // 64*48*2  = 6144
    float*  dposs = (float*)(smem + 151040);      // 2048
    float*  dnegs = (float*)(smem + 153088);      // 2048
    float*  redp  = (float*)(smem + 155136);      // 1024
    float*  redn  = (float*)(smem + 156160);      // 1024
    float*  Mp    = (float*)(smem + 157184);      // 128
    float*  Mn    = (float*)(smem + 157312);      // 128
    float*  Sp    = (float*)(smem + 157440);      // 128
    float*  Sn    = (float*)(smem + 157568);      // 128  -> total 157696 B

    const int t    = threadIdx.x;
    const int lane = t & 31;
    const int w    = t >> 5;
    const int hf   = lane >> 4;
    const int l16  = lane & 15;

    const int qblk = blockIdx.x & 15;
    const int h    = (blockIdx.x >> 4) % Hm;
    const int b    = blockIdx.x / (16 * Hm);
    const int q0   = qblk * 32;

    const int tm = (w & 1) * 16;                // wave tile inside 32x64
    const int tn = (w >> 1) * 16;

    for (int i = t; i < Lm; i += 256) {         // diagonals of W_pos / W_neg
        dposs[i] = Wpos[(size_t)i * Lm + i];
        dnegs[i] = Wneg[(size_t)i * Lm + i];
    }
    {                                           // Q tile 32x64 -> LDS (async)
        int r = t >> 3, c8 = (t & 7) << 3;
        g2l_copy16(&Qs[r * 72 + c8],
                   &qproj[((size_t)(b * Lm + q0 + r)) * Dm + h * DK + c8]);
    }
    g2l_join();
    __syncthreads();

    // ---- Phase 1: S = (q . k^T) / temp, 32x512 panel into LDS ----
    for (int kblk = 0; kblk < 8; ++kblk) {
#pragma unroll
        for (int it = 0; it < 2; ++it) {        // 64x64 K tile, direct copy
            int i = t + it * 256;
            int r = i >> 3, c8 = (i & 7) << 3;
            g2l_copy16(&Ks[r * 72 + c8],
                       &kproj[((size_t)(b * Lm + kblk * 64 + r)) * Dm + h * DK + c8]);
        }
        g2l_join();
        __syncthreads();

        v8f s = {};
        int m = tm + l16;
        int n = tn + l16;
#pragma unroll
        for (int kc = 0; kc < DK; kc += 32) {
            FragBF a, bf;
            a.h[0]  = *(const v8bf*)&Qs[m * 72 + kc + hf * 8];
            a.h[1]  = *(const v8bf*)&Qs[m * 72 + kc + 16 + hf * 8];
            bf.h[0] = *(const v8bf*)&Ks[n * 72 + kc + hf * 8];
            bf.h[1] = *(const v8bf*)&Ks[n * 72 + kc + 16 + hf * 8];
            s = __builtin_amdgcn_wmma_f32_16x16x32_bf16(false, a.v, false, bf.v, (short)0, s, false, false);
        }
#pragma unroll
        for (int vr = 0; vr < 8; ++vr)
            Sf[(tm + vr + 8 * hf) * Lm + kblk * 64 + tn + l16] = s[vr] * INV_TEMP;
        __syncthreads();
    }

    // ---- Phase 2: dual softmax over keys (8 threads per row) ----
    {
        int r = t >> 3, seg = t & 7;
        const float* srow = &Sf[r * Lm];
        float mp = -1e30f, mn = -1e30f;
        for (int j = 0; j < 64; ++j) {
            int c = seg * 64 + j;
            float sv = srow[c];
            mp = fmaxf(mp,  sv * dposs[c]);
            mn = fmaxf(mn, -sv * dnegs[c]);
        }
        redp[r * 8 + seg] = mp; redn[r * 8 + seg] = mn;
        __syncthreads();
        if (t < 32) {
            float m1 = -1e30f, m2 = -1e30f;
            for (int j = 0; j < 8; ++j) {
                m1 = fmaxf(m1, redp[t * 8 + j]);
                m2 = fmaxf(m2, redn[t * 8 + j]);
            }
            Mp[t] = m1; Mn[t] = m2;
        }
        __syncthreads();
        float mprow = Mp[r], mnrow = Mn[r];
        float sp = 0.0f, sn = 0.0f;
        for (int j = 0; j < 64; ++j) {
            int c = seg * 64 + j;
            float sv = srow[c];
            float ep = __expf( sv * dposs[c] - mprow);
            float en = __expf(-sv * dnegs[c] - mnrow);
            Pp[r * Lm + c] = (__bf16)ep;
            Pn[r * Lm + c] = (__bf16)en;
            sp += ep; sn += en;
        }
        redp[r * 8 + seg] = sp; redn[r * 8 + seg] = sn;
        __syncthreads();
        if (t < 32) {
            float s1 = 0.0f, s2 = 0.0f;
            for (int j = 0; j < 8; ++j) { s1 += redp[t * 8 + j]; s2 += redn[t * 8 + j]; }
            Sp[t] = s1; Sn[t] = s2;
        }
        __syncthreads();
    }

    // ---- Phase 3: O_pos = P_pos @ V, O_neg = P_neg @ V (K = 512 keys) ----
    v8f accp = {}, accn = {};
    for (int kc = 0; kc < Lm; kc += 32) {
        {   // stage V chunk transposed: Vs[n][kk] = v[kc+kk][n]
            int kk = t >> 3, n8 = (t & 7) << 3;
            v8bf vv = *(const v8bf*)&vproj[((size_t)(b * Lm + kc + kk)) * Dm + h * DK + n8];
#pragma unroll
            for (int j = 0; j < 8; ++j) Vs[(n8 + j) * 48 + kk] = vv[j];
        }
        __syncthreads();

        FragBF ap, an, bv;
        int m = tm + l16;
        int n = tn + l16;
        ap.h[0] = *(const v8bf*)&Pp[m * Lm + kc + hf * 8];
        ap.h[1] = *(const v8bf*)&Pp[m * Lm + kc + 16 + hf * 8];
        an.h[0] = *(const v8bf*)&Pn[m * Lm + kc + hf * 8];
        an.h[1] = *(const v8bf*)&Pn[m * Lm + kc + 16 + hf * 8];
        bv.h[0] = *(const v8bf*)&Vs[n * 48 + hf * 8];
        bv.h[1] = *(const v8bf*)&Vs[n * 48 + 16 + hf * 8];

        accp = __builtin_amdgcn_wmma_f32_16x16x32_bf16(false, ap.v, false, bv.v, (short)0, accp, false, false);
        accn = __builtin_amdgcn_wmma_f32_16x16x32_bf16(false, an.v, false, bv.v, (short)0, accn, false, false);
        __syncthreads();
    }

    // Epilogue: divide by softmax sums, negate the negative stream
#pragma unroll
    for (int vr = 0; vr < 8; ++vr) {
        int row = tm + vr + 8 * hf;
        int col = tn + l16;
        float op =  accp[vr] / Sp[row];
        float on = -(accn[vr] / Sn[row]);
        size_t gi = ((size_t)(b * Lm + q0 + row)) * Dm + h * DK + col;
        Vp[gi] = (__bf16)op;
        Vn[gi] = (__bf16)on;
    }
}

// ---------------------------------------------------------------------------
// Host-side orchestration
// ---------------------------------------------------------------------------
extern "C" void kernel_launch(void* const* d_in, const int* in_sizes, int n_in,
                              void* d_out, int out_size, void* d_ws, size_t ws_size,
                              hipStream_t stream) {
    const float* Q    = (const float*)d_in[0];
    const float* Kin  = (const float*)d_in[1];
    const float* Vin  = (const float*)d_in[2];
    const float* W_q  = (const float*)d_in[3];
    const float* W_k  = (const float*)d_in[4];
    const float* W_v  = (const float*)d_in[5];
    const float* Wpos = (const float*)d_in[6];
    const float* Wneg = (const float*)d_in[7];
    const float* W_o  = (const float*)d_in[8];
    const float* W_o2 = (const float*)d_in[9];
    const float* w1   = (const float*)d_in[10];
    const float* b1   = (const float*)d_in[11];
    const float* w2   = (const float*)d_in[12];
    const float* b2   = (const float*)d_in[13];
    float* out = (float*)d_out;

    const size_t ACT = (size_t)Mtot * Dm * 2;   // bf16 activation buffer
    const size_t WB  = (size_t)Dm * Dm * 2;     // bf16 weight buffer
    char* ws = (char*)d_ws;
    size_t off = 0;
    auto carve = [&](size_t n) { char* p = ws + off; off += (n + 255) & ~(size_t)255; return p; };

    __bf16* A0 = (__bf16*)carve(ACT);  // Q bf16   -> later V_pos
    __bf16* A1 = (__bf16*)carve(ACT);  // K bf16   -> later V_neg
    __bf16* A2 = (__bf16*)carve(ACT);  // V bf16   -> later H_pos
    __bf16* A3 = (__bf16*)carve(ACT);  // q proj   -> later X_pos bf16
    __bf16* A4 = (__bf16*)carve(ACT);  // k proj   -> later X_neg bf16
    __bf16* A5 = (__bf16*)carve(ACT);  // v proj   -> later H_neg
    __bf16* Wqb  = (__bf16*)carve(WB);
    __bf16* Wkb  = (__bf16*)carve(WB);
    __bf16* Wvb  = (__bf16*)carve(WB);
    __bf16* Wob  = (__bf16*)carve(WB);
    __bf16* Wo2b = (__bf16*)carve(WB);
    __bf16* W1b  = (__bf16*)carve(WB);
    __bf16* W2b  = (__bf16*)carve(WB);
    float*  Xp   = (float*)carve((size_t)Mtot * Dm * 4);
    float*  Xn   = (float*)carve((size_t)Mtot * Dm * 4);

    const int nAct4 = Mtot * Dm / 4;
    const int nW4   = Dm * Dm / 4;
    const dim3 cb(256);
    auto cgrid = [](int n) { return dim3((n + 255) / 256); };

    // 1) bf16 conversions
    cvt_f32_bf16<<<cgrid(nAct4), cb, 0, stream>>>(Q,   A0, nAct4);
    cvt_f32_bf16<<<cgrid(nAct4), cb, 0, stream>>>(Kin, A1, nAct4);
    cvt_f32_bf16<<<cgrid(nAct4), cb, 0, stream>>>(Vin, A2, nAct4);
    cvt_f32_bf16<<<cgrid(nW4), cb, 0, stream>>>(W_q,  Wqb,  nW4);
    cvt_f32_bf16<<<cgrid(nW4), cb, 0, stream>>>(W_k,  Wkb,  nW4);
    cvt_f32_bf16<<<cgrid(nW4), cb, 0, stream>>>(W_v,  Wvb,  nW4);
    cvt_f32_bf16<<<cgrid(nW4), cb, 0, stream>>>(W_o,  Wob,  nW4);
    cvt_f32_bf16<<<cgrid(nW4), cb, 0, stream>>>(W_o2, Wo2b, nW4);
    cvt_f32_bf16<<<cgrid(nW4), cb, 0, stream>>>(w1,   W1b,  nW4);
    cvt_f32_bf16<<<cgrid(nW4), cb, 0, stream>>>(w2,   W2b,  nW4);

    const dim3 ggrid(Mtot / 128, Dm / 64);   // (32, 12)

    // 2) projections: q/k/v = X @ W  (bf16 out)
    gemm_bf16_wmma<0><<<ggrid, cb, 0, stream>>>(A0, Wqb, nullptr, nullptr, nullptr, A3, Mtot, Dm, Dm);
    gemm_bf16_wmma<0><<<ggrid, cb, 0, stream>>>(A1, Wkb, nullptr, nullptr, nullptr, A4, Mtot, Dm, Dm);
    gemm_bf16_wmma<0><<<ggrid, cb, 0, stream>>>(A2, Wvb, nullptr, nullptr, nullptr, A5, Mtot, Dm, Dm);

    // 3) fused dual-softmax attention -> V_pos (A0), V_neg (A1)
    const int ATTN_SMEM = 157696;
    (void)hipFuncSetAttribute((const void*)attn_dual_softmax,
                              hipFuncAttributeMaxDynamicSharedMemorySize, ATTN_SMEM);
    attn_dual_softmax<<<dim3(Bm * Hm * (Lm / 32)), cb, ATTN_SMEM, stream>>>(
        A3, A4, A5, Wpos, Wneg, A0, A1);

    // 4) output projections + residual: X = Q + Vx @ W_o  (f32 + bf16 copies)
    gemm_bf16_wmma<2><<<ggrid, cb, 0, stream>>>(A0, Wob,  nullptr, Q, Xp, A3, Mtot, Dm, Dm);
    gemm_bf16_wmma<2><<<ggrid, cb, 0, stream>>>(A1, Wo2b, nullptr, Q, Xn, A4, Mtot, Dm, Dm);

    // 5) FFN layer 1: H = relu(X @ w1 + b1)  (bf16 out)
    gemm_bf16_wmma<1><<<ggrid, cb, 0, stream>>>(A3, W1b, b1, nullptr, nullptr, A2, Mtot, Dm, Dm);
    gemm_bf16_wmma<1><<<ggrid, cb, 0, stream>>>(A4, W1b, b1, nullptr, nullptr, A5, Mtot, Dm, Dm);

    // 6) FFN layer 2 + residual: out = H @ w2 + b2 + X  (f32 -> d_out)
    gemm_bf16_wmma<3><<<ggrid, cb, 0, stream>>>(A2, W2b, b2, Xp, out,            nullptr, Mtot, Dm, Dm);
    gemm_bf16_wmma<3><<<ggrid, cb, 0, stream>>>(A5, W2b, b2, Xn, out + (size_t)Mtot * Dm, nullptr, Mtot, Dm, Dm);
}
